// MultiHeadAttention_37039797961289
// MI455X (gfx1250) — matrix operand order repaired and checked by reading, not compiled, not run on hardware
//
#include <hip/hip_runtime.h>
#include <hip/hip_bf16.h>

#define D_  768
#define H_  12
#define HD_ 64
#define S_  4096

typedef __attribute__((ext_vector_type(16))) _Float16 v16h;
typedef __attribute__((ext_vector_type(8)))  _Float16 v8h;
typedef __attribute__((ext_vector_type(8)))  float    v8f;

// ---- WMMA wrapper: D(f32 16x16) = A(f16 16x32) * B(f16 32x16) + C ----
__device__ __forceinline__ v8f wmma16(v16h a, v16h b, v8f c) {
  return __builtin_amdgcn_wmma_f32_16x16x32_f16(false, a, false, b, (short)0, c, false, false);
}

// A-fragment (16x32 f16) from row-major matrix, row stride ld (halves).
// Lane layout (ISA 7.12.2): lane holds row M = lane&15; element e -> K = (e>>3)*16 + half*8 + (e&7).
__device__ __forceinline__ v16h load_a(const _Float16* base, int ld, int row, int kbase, int hl) {
  const _Float16* p = base + (size_t)row * ld + kbase + hl * 8;
  v8h lo = *(const v8h*)p;
  v8h hi = *(const v8h*)(p + 16);
  v16h a;
#pragma unroll
  for (int i = 0; i < 8; i++) { a[i] = lo[i]; a[i + 8] = hi[i]; }
  return a;
}

// B-fragment (32x16 f16) from "transposed" storage: row n holds contiguous K, stride ld.
// Lane layout: lane holds column N = lane&15; element e -> K = kbase + (e>>3)*16 + half*8 + (e&7).
__device__ __forceinline__ v16h load_b(const _Float16* base, int ld, int nrow, int kbase, int hl) {
  const _Float16* p = base + (size_t)nrow * ld + kbase + hl * 8;
  v8h lo = *(const v8h*)p;
  v8h hi = *(const v8h*)(p + 16);
  v16h b;
#pragma unroll
  for (int i = 0; i < 8; i++) { b[i] = lo[i]; b[i + 8] = hi[i]; }
  return b;
}

// xor-shuffle reductions across the 16 lanes of a half-wave (wave32)
__device__ __forceinline__ float rmax16(float v) {
#pragma unroll
  for (int m = 1; m <= 8; m <<= 1) v = fmaxf(v, __shfl_xor(v, m, 32));
  return v;
}
__device__ __forceinline__ float rsum16(float v) {
#pragma unroll
  for (int m = 1; m <= 8; m <<= 1) v += __shfl_xor(v, m, 32);
  return v;
}

// ---------------- conversion / layout kernels ----------------
__global__ __launch_bounds__(256) void cvt_f16(const float* __restrict__ src,
                                               _Float16* __restrict__ dst, int n) {
  int i = blockIdx.x * 256 + threadIdx.x;
  if (i < n) dst[i] = (_Float16)src[i];
}

// [H, D, HD] f32 -> [H, HD, D] f16  (per-head transpose)
__global__ __launch_bounds__(256) void cvt_wheads(const float* __restrict__ src,
                                                  _Float16* __restrict__ dst) {
  int i = blockIdx.x * 256 + threadIdx.x;
  if (i >= H_ * D_ * HD_) return;
  int h = i / (D_ * HD_);
  int r = i - h * (D_ * HD_);
  int d = r / HD_;
  int e = r - d * HD_;
  dst[((size_t)h * HD_ + e) * D_ + d] = (_Float16)src[i];
}

// [D, D] f32 -> [D, D] f16 transposed
__global__ __launch_bounds__(256) void cvt_wo(const float* __restrict__ src,
                                              _Float16* __restrict__ dst) {
  int i = blockIdx.x * 256 + threadIdx.x;
  if (i >= D_ * D_) return;
  int k = i / D_, n = i - k * D_;
  dst[(size_t)n * D_ + k] = (_Float16)src[i];
}

// ---------------- QKV projection ----------------
// one wave per (16-row tile, head). Q scaled by 1/sqrt(HD)=0.125 at write-out.
__global__ __launch_bounds__(256) void qkv_kernel(
    const _Float16* __restrict__ xh, const _Float16* __restrict__ wqT,
    const _Float16* __restrict__ wkT, const _Float16* __restrict__ wvT,
    const float* __restrict__ bq, const float* __restrict__ bk, const float* __restrict__ bv,
    _Float16* __restrict__ Qh, _Float16* __restrict__ Kh, _Float16* __restrict__ VhT) {
  int lane = threadIdx.x & 31, wave = threadIdx.x >> 5;
  int wid  = blockIdx.x * 8 + wave;
  int h    = wid >> 8;          // 0..11
  int tile = wid & 255;         // 0..255
  int hl = lane >> 4, lrow = lane & 15;
  const _Float16* wq = wqT + (size_t)h * HD_ * D_;
  const _Float16* wk = wkT + (size_t)h * HD_ * D_;
  const _Float16* wv = wvT + (size_t)h * HD_ * D_;
  int row = tile * 16 + lrow;
  v8f aq[4], ak[4], av[4];
#pragma unroll
  for (int t = 0; t < 4; t++) { aq[t] = (v8f){}; ak[t] = (v8f){}; av[t] = (v8f){}; }
  for (int kb = 0; kb < D_; kb += 32) {
    v16h a = load_a(xh, D_, row, kb, hl);
#pragma unroll
    for (int t = 0; t < 4; t++) {
      aq[t] = wmma16(a, load_b(wq, D_, t * 16 + lrow, kb, hl), aq[t]);
      ak[t] = wmma16(a, load_b(wk, D_, t * 16 + lrow, kb, hl), ak[t]);
      av[t] = wmma16(a, load_b(wv, D_, t * 16 + lrow, kb, hl), av[t]);
    }
  }
#pragma unroll
  for (int r = 0; r < 8; r++) {
    int m = hl * 8 + r;
    int srow = tile * 16 + m;
#pragma unroll
    for (int t = 0; t < 4; t++) {
      int n = t * 16 + lrow;
      Qh[((size_t)h * S_ + srow) * HD_ + n] = (_Float16)((aq[t][r] + bq[h * HD_ + n]) * 0.125f);
      Kh[((size_t)h * S_ + srow) * HD_ + n] = (_Float16)(ak[t][r] + bk[h * HD_ + n]);
      VhT[(size_t)h * HD_ * S_ + (size_t)n * S_ + srow] = (_Float16)(av[t][r] + bv[h * HD_ + n]);
    }
  }
}

// ---------------- flash attention ----------------
// one wave per 16-query tile per head; online softmax over 32-key chunks.
__global__ __launch_bounds__(256) void attn_kernel(
    const _Float16* __restrict__ Qh, const _Float16* __restrict__ Kh,
    const _Float16* __restrict__ VhT, _Float16* __restrict__ Ch) {
  __shared__ _Float16 Pbuf[8 * 16 * 32];   // per-wave 16x32 P tile
  int lane = threadIdx.x & 31, wave = threadIdx.x >> 5;
  int wid  = blockIdx.x * 8 + wave;
  int h    = wid >> 8;
  int tile = wid & 255;
  int hl = lane >> 4, lrow = lane & 15;
  const _Float16* Qb = Qh  + (size_t)h * S_ * HD_;
  const _Float16* Kb = Kh  + (size_t)h * S_ * HD_;
  const _Float16* Vb = VhT + (size_t)h * HD_ * S_;
  _Float16* P = Pbuf + wave * (16 * 32);

  int qrow = tile * 16 + lrow;
  v16h aq0 = load_a(Qb, HD_, qrow, 0, hl);
  v16h aq1 = load_a(Qb, HD_, qrow, 32, hl);

  float mrun[8], lrun[8];
  v8f accO[4];
#pragma unroll
  for (int r = 0; r < 8; r++) { mrun[r] = -1e30f; lrun[r] = 0.f; }
#pragma unroll
  for (int t = 0; t < 4; t++) accO[t] = (v8f){};

  for (int cb = 0; cb < S_; cb += 32) {
    // scores: 16 queries x 32 keys
    v8f s0 = (v8f){}, s1 = (v8f){};
    s0 = wmma16(aq0, load_b(Kb, HD_, cb + lrow, 0, hl), s0);
    s0 = wmma16(aq1, load_b(Kb, HD_, cb + lrow, 32, hl), s0);
    s1 = wmma16(aq0, load_b(Kb, HD_, cb + 16 + lrow, 0, hl), s1);
    s1 = wmma16(aq1, load_b(Kb, HD_, cb + 16 + lrow, 32, hl), s1);
    // online softmax (row stats via 16-lane xor reductions)
#pragma unroll
    for (int r = 0; r < 8; r++) {
      float cm = rmax16(fmaxf(s0[r], s1[r]));
      float nm = fmaxf(mrun[r], cm);
      float alpha = __expf(mrun[r] - nm);
      mrun[r] = nm;
      float p0 = __expf(s0[r] - nm);
      float p1 = __expf(s1[r] - nm);
      lrun[r] = lrun[r] * alpha + rsum16(p0 + p1);
#pragma unroll
      for (int t = 0; t < 4; t++) accO[t][r] *= alpha;
      int m = hl * 8 + r;
      P[m * 32 + lrow]      = (_Float16)p0;
      P[m * 32 + 16 + lrow] = (_Float16)p1;
    }
    __builtin_amdgcn_wave_barrier();      // keep LDS store->load order (HW is in-order per wave)
    v16h ap = load_a(P, 32, lrow, 0, hl); // reload P in A-fragment layout
#pragma unroll
    for (int t = 0; t < 4; t++)
      accO[t] = wmma16(ap, load_b(Vb, S_, t * 16 + lrow, cb, hl), accO[t]);
  }
  // normalize and store into concat layout [S, D] f16
#pragma unroll
  for (int r = 0; r < 8; r++) {
    float inv = 1.0f / lrun[r];
    int m = hl * 8 + r;
    int row = tile * 16 + m;
#pragma unroll
    for (int t = 0; t < 4; t++)
      Ch[(size_t)row * D_ + h * HD_ + t * 16 + lrow] = (_Float16)(accO[t][r] * inv);
  }
}

// ---------------- output projection ----------------
__global__ __launch_bounds__(256) void outproj_kernel(
    const _Float16* __restrict__ Ch, const _Float16* __restrict__ woT,
    const float* __restrict__ bo, float* __restrict__ out) {
  int lane = threadIdx.x & 31, wave = threadIdx.x >> 5;
  int wid  = blockIdx.x * 8 + wave;
  int ct = wid >> 8;        // 0..11 column tiles of 64
  int rt = wid & 255;       // 0..255 row tiles of 16
  int hl = lane >> 4, lrow = lane & 15;
  int row = rt * 16 + lrow;
  v8f acc[4];
#pragma unroll
  for (int t = 0; t < 4; t++) acc[t] = (v8f){};
  for (int kb = 0; kb < D_; kb += 32) {
    v16h a = load_a(Ch, D_, row, kb, hl);
#pragma unroll
    for (int t = 0; t < 4; t++)
      acc[t] = wmma16(a, load_b(woT, D_, ct * 64 + t * 16 + lrow, kb, hl), acc[t]);
  }
#pragma unroll
  for (int r = 0; r < 8; r++) {
    int m = hl * 8 + r;
    int orow = rt * 16 + m;
#pragma unroll
    for (int t = 0; t < 4; t++) {
      int n = ct * 64 + t * 16 + lrow;
      out[(size_t)orow * D_ + n] = acc[t][r] + bo[n];
    }
  }
}

extern "C" void kernel_launch(void* const* d_in, const int* in_sizes, int n_in,
                              void* d_out, int out_size, void* d_ws, size_t ws_size,
                              hipStream_t stream) {
  const float* x  = (const float*)d_in[0];
  const float* Wq = (const float*)d_in[1];
  const float* Wk = (const float*)d_in[2];
  const float* Wv = (const float*)d_in[3];
  const float* bq = (const float*)d_in[4];
  const float* bk = (const float*)d_in[5];
  const float* bv = (const float*)d_in[6];
  const float* Wo = (const float*)d_in[7];
  const float* bo = (const float*)d_in[8];
  float* out = (float*)d_out;

  _Float16* ws = (_Float16*)d_ws;
  size_t off = 0;
  _Float16* xh  = ws + off; off += (size_t)S_ * D_;
  _Float16* wqT = ws + off; off += (size_t)H_ * HD_ * D_;
  _Float16* wkT = ws + off; off += (size_t)H_ * HD_ * D_;
  _Float16* wvT = ws + off; off += (size_t)H_ * HD_ * D_;
  _Float16* woT = ws + off; off += (size_t)D_ * D_;
  _Float16* Qh  = ws + off; off += (size_t)H_ * S_ * HD_;
  _Float16* Kh  = ws + off; off += (size_t)H_ * S_ * HD_;
  _Float16* VhT = ws + off; off += (size_t)H_ * HD_ * S_;
  _Float16* Ch  = ws + off; off += (size_t)S_ * D_;

  cvt_f16<<<(S_ * D_ + 255) / 256, 256, 0, stream>>>(x, xh, S_ * D_);
  cvt_wheads<<<(H_ * D_ * HD_ + 255) / 256, 256, 0, stream>>>(Wq, wqT);
  cvt_wheads<<<(H_ * D_ * HD_ + 255) / 256, 256, 0, stream>>>(Wk, wkT);
  cvt_wheads<<<(H_ * D_ * HD_ + 255) / 256, 256, 0, stream>>>(Wv, wvT);
  cvt_wo<<<(D_ * D_ + 255) / 256, 256, 0, stream>>>(Wo, woT);

  // 12 heads * 256 row tiles = 3072 waves = 384 blocks of 8 waves
  qkv_kernel<<<384, 256, 0, stream>>>(xh, wqT, wkT, wvT, bq, bk, bv, Qh, Kh, VhT);
  attn_kernel<<<384, 256, 0, stream>>>(Qh, Kh, VhT, Ch);
  outproj_kernel<<<384, 256, 0, stream>>>(Ch, woT, bo, out);
}
